// SAGE_352187318570
// MI455X (gfx1250) — matrix operand order repaired and checked by reading, not compiled, not run on hardware
//
#include <hip/hip_runtime.h>

typedef __attribute__((ext_vector_type(16))) _Float16 v16h;
typedef __attribute__((ext_vector_type(8)))  _Float16 v8h;
typedef __attribute__((ext_vector_type(8)))  float    v8f;

#define N_NODES 50000
#define N_EDGES 800000
#define EDIM    64
#define NDIM_IN 64
#define DOUT    128            // node output dim, both layers
#define EK      (2*DOUT+EDIM)  // 320, edge-linear K

// ---------------------------------------------------------------- utilities

__global__ void k_convert_f16(const float* __restrict__ s, _Float16* __restrict__ d, int n) {
    int i = blockIdx.x * 256 + threadIdx.x;
    if (i < n) d[i] = (_Float16)s[i];
}

__global__ void k_deg(const int* __restrict__ dst, float* __restrict__ deg) {
    int e = blockIdx.x * 256 + threadIdx.x;
    if (e < N_EDGES) atomicAdd(&deg[dst[e]], 1.0f);
}

// one thread per (edge, 4-float chunk): 800K*16 threads, f32 atomics hit L2
__global__ void k_scatter(const float* ef, const int* __restrict__ dst, float* msum) {
    int idx = blockIdx.x * 256 + threadIdx.x;
    if (idx >= N_EDGES * 16) return;
    int e = idx >> 4;
    int q = (idx & 15) << 2;
    float4 v = *(const float4*)(ef + (size_t)e * EDIM + q);
    float* o = msum + (size_t)dst[e] * EDIM + q;
    atomicAdd(o + 0, v.x); atomicAdd(o + 1, v.y);
    atomicAdd(o + 2, v.z); atomicAdd(o + 3, v.w);
}

// --------------------------------------------------------- WMMA frag loads

// A (16x32 f16) from LDS row-major tile; this code's contiguous-K layout (not the ISA table order):
// lanes 0-15: M=lane, elems 0-15 = K k0..k0+15
// lanes16-31: M=lane-16, elems shifted by +16 in K
__device__ __forceinline__ v16h load_a(const _Float16* arow, int k0) {
    v8h lo = *(const v8h*)(arow + k0);
    v8h hi = *(const v8h*)(arow + k0 + 8);
    return __builtin_shufflevector(lo, hi, 0,1,2,3,4,5,6,7,8,9,10,11,12,13,14,15);
}

// ------------------------------------------------------------- node kernel
// 256 threads = 8 waves; block handles 16 nodes; wave w owns out cols [16w,16w+16)
template <int KFEAT, bool FEAT16, bool WRITE32>
__global__ void __launch_bounds__(256)
k_node(const void* __restrict__ featp, const float* __restrict__ msum,
       const float* __restrict__ deg, const _Float16* __restrict__ Wh,
       const float* __restrict__ bias, _Float16* __restrict__ h16,
       float* __restrict__ h32)
{
    constexpr int K = KFEAT + EDIM;            // 128 (L1) or 192 (L2)
    __shared__ __align__(16) _Float16 sA[16 * K];

    const int tid   = threadIdx.x;
    const int nbase = blockIdx.x * 16;

    // stage A = concat(feat, msum/deg) as f16
    for (int el = tid; el < 16 * K; el += 256) {
        int row = el / K, c = el - row * K;
        int node = nbase + row;
        _Float16 v;
        if (c < KFEAT) {
            if (FEAT16) v = ((const _Float16*)featp)[(size_t)node * KFEAT + c];
            else        v = (_Float16)(((const float*)featp)[(size_t)node * KFEAT + c]);
        } else {
            float dg = deg[node];
            v = (_Float16)(msum[(size_t)node * EDIM + (c - KFEAT)] / fmaxf(dg, 1.0f));
        }
        sA[el] = v;
    }
    __syncthreads();

    const int lane = tid & 31, wave = tid >> 5;
    const int ncol = (wave << 4) + (lane & 15);
    const int hik  = (lane & 16) ? 16 : 0;
    const _Float16* wrow = Wh + (size_t)ncol * K + hik;               // B frag base
    const _Float16* arow = sA + (lane & 15) * K + ((lane & 16) ? 16 : 0);

    v8f acc = {};
#pragma unroll
    for (int k0 = 0; k0 < K; k0 += 32) {
        v16h a = load_a(arow, k0);
        v16h b = *(const v16h*)(wrow + k0);
        acc = __builtin_amdgcn_wmma_f32_16x16x32_f16(false, a, false, b,
                                                     (short)0, acc, false, false);
    }

    float bv = bias[ncol];
    int mbase = (lane & 16) ? 8 : 0;
#pragma unroll
    for (int r = 0; r < 8; r++) {
        int node = nbase + mbase + r;
        float y = fmaxf(acc[r] + bv, 0.0f);
        h16[(size_t)node * DOUT + ncol] = (_Float16)y;
        if (WRITE32) h32[(size_t)node * DOUT + ncol] = y;
    }
}

// ------------------------------------------------------------- edge kernel
// 128 threads = 4 waves; block handles 16 edges; A = [h[src] | h[dst] | ef]
// NOTE: ef32 may alias eout (layer 2 in-place): reads finish before the
// barrier, writes happen after — block-private rows, so no hazard.
__global__ void __launch_bounds__(128)
k_edge(const _Float16* __restrict__ h16, const float* ef32,
       const int* __restrict__ src, const int* __restrict__ dst,
       const _Float16* __restrict__ eWh, const float* __restrict__ ebias,
       float* eout)
{
    __shared__ __align__(16) _Float16 sA[16 * EK];
    __shared__ int s_src[16], s_dst[16];

    const int tid   = threadIdx.x;
    const int ebase = blockIdx.x * 16;
    if (tid < 16)      s_src[tid]      = src[ebase + tid];
    else if (tid < 32) s_dst[tid - 16] = dst[ebase + tid - 16];
    __syncthreads();

    for (int el = tid; el < 16 * EK; el += 128) {
        int row = el / EK, c = el - row * EK;
        _Float16 v;
        if (c < DOUT)          v = h16[(size_t)s_src[row] * DOUT + c];
        else if (c < 2 * DOUT) v = h16[(size_t)s_dst[row] * DOUT + (c - DOUT)];
        else v = (_Float16)ef32[(size_t)(ebase + row) * EDIM + (c - 2 * DOUT)];
        sA[el] = v;
    }
    __syncthreads();

    const int lane = tid & 31, wave = tid >> 5;
    const int ncol = (wave << 4) + (lane & 15);                        // 0..63
    const _Float16* wrow = eWh + (size_t)ncol * EK + ((lane & 16) ? 16 : 0);
    const _Float16* arow = sA + (lane & 15) * EK + ((lane & 16) ? 16 : 0);

    v8f acc = {};
#pragma unroll
    for (int k0 = 0; k0 < EK; k0 += 32) {
        v16h a = load_a(arow, k0);
        v16h b = *(const v16h*)(wrow + k0);
        acc = __builtin_amdgcn_wmma_f32_16x16x32_f16(false, a, false, b,
                                                     (short)0, acc, false, false);
    }

    float bv = ebias[ncol];
    int mbase = (lane & 16) ? 8 : 0;
#pragma unroll
    for (int r = 0; r < 8; r++) {
        int e = ebase + mbase + r;
        eout[(size_t)e * EDIM + ncol] = fmaxf(acc[r] + bv, 0.0f);
    }
}

// ---------------------------------------------------------------- launcher

extern "C" void kernel_launch(void* const* d_in, const int* in_sizes, int n_in,
                              void* d_out, int out_size, void* d_ws, size_t ws_size,
                              hipStream_t stream)
{
    const float* nf  = (const float*)d_in[0];
    const float* ef  = (const float*)d_in[1];
    const int*   src = (const int*)d_in[2];
    const int*   dst = (const int*)d_in[3];
    const float* W1  = (const float*)d_in[4];
    const float* b1  = (const float*)d_in[5];
    const float* eW1 = (const float*)d_in[6];
    const float* eb1 = (const float*)d_in[7];
    const float* W2  = (const float*)d_in[8];
    const float* b2  = (const float*)d_in[9];
    const float* eW2 = (const float*)d_in[10];
    const float* eb2 = (const float*)d_in[11];

    float* out_h = (float*)d_out;                      // [N, 128]
    float* out_e = out_h + (size_t)N_NODES * DOUT;     // [E, 64]  (also ef1 scratch)

    char* ws = (char*)d_ws;
    size_t off = 0;
    auto walloc = [&](size_t bytes) -> char* {
        char* p = ws + off;
        off = (off + bytes + 255) & ~(size_t)255;
        return p;
    };
    float*    msum = (float*)   walloc((size_t)N_NODES * EDIM * 4);
    float*    deg  = (float*)   walloc((size_t)N_NODES * 4);
    _Float16* h1   = (_Float16*)walloc((size_t)N_NODES * DOUT * 2);
    _Float16* h2   = (_Float16*)walloc((size_t)N_NODES * DOUT * 2);
    _Float16* W1h  = (_Float16*)walloc(128 * 128 * 2);
    _Float16* eW1h = (_Float16*)walloc(64 * EK * 2);
    _Float16* W2h  = (_Float16*)walloc(128 * 192 * 2);
    _Float16* eW2h = (_Float16*)walloc(64 * EK * 2);
    (void)ws_size; (void)in_sizes; (void)n_in; (void)out_size;

    // f16 weight tables (K-contiguous rows -> aligned B-fragment loads)
    k_convert_f16<<<(128 * 128 + 255) / 256, 256, 0, stream>>>(W1, W1h, 128 * 128);
    k_convert_f16<<<(64 * EK + 255) / 256, 256, 0, stream>>>(eW1, eW1h, 64 * EK);
    k_convert_f16<<<(128 * 192 + 255) / 256, 256, 0, stream>>>(W2, W2h, 128 * 192);
    k_convert_f16<<<(64 * EK + 255) / 256, 256, 0, stream>>>(eW2, eW2h, 64 * EK);

    // ---- layer 1
    hipMemsetAsync(msum, 0, (size_t)N_NODES * EDIM * 4, stream);
    hipMemsetAsync(deg, 0, (size_t)N_NODES * 4, stream);
    k_deg<<<(N_EDGES + 255) / 256, 256, 0, stream>>>(dst, deg);
    k_scatter<<<(N_EDGES * 16 + 255) / 256, 256, 0, stream>>>(ef, dst, msum);
    k_node<NDIM_IN, false, false><<<N_NODES / 16, 256, 0, stream>>>(
        nf, msum, deg, W1h, b1, h1, nullptr);
    k_edge<<<N_EDGES / 16, 128, 0, stream>>>(h1, ef, src, dst, eW1h, eb1, out_e);

    // ---- layer 2 (deg unchanged; ef1 lives in out_e as f32)
    hipMemsetAsync(msum, 0, (size_t)N_NODES * EDIM * 4, stream);
    k_scatter<<<(N_EDGES * 16 + 255) / 256, 256, 0, stream>>>(out_e, dst, msum);
    k_node<DOUT, true, true><<<N_NODES / 16, 256, 0, stream>>>(
        h1, msum, deg, W2h, b2, h2, out_h);
    k_edge<<<N_EDGES / 16, 128, 0, stream>>>(h2, out_e, src, dst, eW2h, eb2, out_e);
}
